// TrajectoryLSTM_45097156608097
// MI455X (gfx1250) — compile-verified, matrix operand order checked by reading, not run on hardware
//
#include <hip/hip_runtime.h>

typedef __attribute__((ext_vector_type(16))) __bf16 v16bf;
typedef __attribute__((ext_vector_type(8)))  float  v8f;

union Frag32 {            // one 16x16x32 bf16 WMMA A or B operand (8 VGPRs)
  v16bf v;
  uint4 q[2];
};

// gfx1250 hardware tanh (shortens the serial pointwise chain); guarded so the
// build cannot fail if the builtin name differs in this toolchain.
#if __has_builtin(__builtin_amdgcn_tanhf)
#define HW_TANH __builtin_amdgcn_tanhf
#elif __has_builtin(__builtin_amdgcn_tanh_f32)
#define HW_TANH __builtin_amdgcn_tanh_f32
#endif

#ifdef HW_TANH
__device__ __forceinline__ float fast_tanh(float x)    { return HW_TANH(x); }
__device__ __forceinline__ float fast_sigmoid(float x) { return __builtin_fmaf(HW_TANH(x * 0.5f), 0.5f, 0.5f); }
#else
__device__ __forceinline__ float fast_sigmoid(float x) { return 1.0f / (1.0f + __expf(-x)); }
__device__ __forceinline__ float fast_tanh(float x)    { return 1.0f - 2.0f / (__expf(2.0f * x) + 1.0f); }
#endif

// ---------------------------------------------------------------------------
// Utility kernels
// ---------------------------------------------------------------------------
__global__ void cvt_f32_bf16_kernel(const float* __restrict__ src, __bf16* __restrict__ dst, int n) {
  for (int i = blockIdx.x * blockDim.x + threadIdx.x; i < n; i += gridDim.x * blockDim.x)
    dst[i] = (__bf16)src[i];
}

__global__ void vec_add_kernel(const float* __restrict__ a, const float* __restrict__ b,
                               float* __restrict__ dst, int n) {
  int i = blockIdx.x * blockDim.x + threadIdx.x;
  if (i < n) dst[i] = a[i] + b[i];
}

// ---------------------------------------------------------------------------
// Persistent fused LSTM layer.
//   xin : [B][T][KIN]  bf16   (layer input)
//   wih : [512][KIN]   bf16   (staged to padded LDS)
//   whh : [512][128]   bf16   (preloaded into per-wave VGPR fragments)
//   bias: [512]        f32    (b_ih + b_hh; WMMA C-init via loop-invariant regs)
//   hout: [B][T][128]  bf16
// grid = B/16 blocks of 256 threads (8 waves); wave w owns gate tiles
// {w, w+8, w+16, w+24}  ->  i/f/g/o for columns [16w,16w+16).
// Time loop unrolled x2 with double-buffered x fragments: step t preloads
// x(t+1) so global latency is hidden behind a full step.
// ---------------------------------------------------------------------------
template <int KIN>
__global__ __launch_bounds__(256)
void lstm_layer_kernel(const __bf16* __restrict__ xin,
                       const __bf16* __restrict__ wih,
                       const __bf16* __restrict__ whh,
                       const float*  __restrict__ bias,
                       __bf16* __restrict__ hout,
                       int T) {
  constexpr int H  = 128;
  constexpr int NG = 4 * H;                 // 512 gate rows
  constexpr int WIH_STRIDE = KIN * 2 + 16;  // padded LDS row (bank-conflict free)
  constexpr int H_STRIDE   = H * 2 + 16;    // 272 bytes
  constexpr int NKT = KIN / 32;

  extern __shared__ char smem[];
  char* s_wih = smem;                          // NG * WIH_STRIDE
  char* s_h   = smem + NG * WIH_STRIDE;        // 2 * 16 * H_STRIDE (double buffer)

  const int tid   = threadIdx.x;
  const int lane  = tid & 31;
  const int wv    = tid >> 5;                  // wave 0..7
  const int ln    = lane & 15;
  const int khalf = (lane & 16) ? 8 : 0;       // 16-bit A/B layout half-split
  const int b0    = blockIdx.x * 16;

  // ---- stage W_ih into padded LDS; zero h double-buffer ----
  {
    constexpr int row_chunks = (KIN * 2) / 16;
    for (int c = tid; c < NG * row_chunks; c += 256) {
      int row = c / row_chunks, col = c % row_chunks;
      uint4 d = ((const uint4*)(wih + (size_t)row * KIN))[col];
      *(uint4*)(s_wih + row * WIH_STRIDE + col * 16) = d;
    }
    for (int i = tid; i < 2 * 16 * H_STRIDE / 4; i += 256)
      ((int*)s_h)[i] = 0;
  }

  // ---- preload W_hh B-fragments (16 frags = 128 VGPRs) + bias C vectors ----
  Frag32 bhh[4][4];
  v8f    bias_v[4];                            // loop-invariant WMMA C-init
#pragma unroll
  for (int j = 0; j < 4; ++j) {
    int ng = (wv + 8 * j) * 16 + ln;           // gate row == WMMA column
    float b = bias[ng];
    bias_v[j] = (v8f){b, b, b, b, b, b, b, b};
    const __bf16* row = whh + (size_t)ng * H;
#pragma unroll
    for (int kt = 0; kt < 4; ++kt) {
      int klo = kt * 32 + khalf;
      bhh[j][kt].q[0] = *(const uint4*)(row + klo);
      bhh[j][kt].q[1] = *(const uint4*)(row + klo + 16);
    }
  }

  v8f c = {0.f, 0.f, 0.f, 0.f, 0.f, 0.f, 0.f, 0.f};   // cell state, 16x16 tile
  const char* xrow = (const char*)(xin + ((size_t)(b0 + ln)) * T * KIN);

  auto load_x = [&](Frag32 (&f)[NKT], int t) {
    const char* xbase = xrow + (size_t)t * KIN * 2;
#pragma unroll
    for (int kt = 0; kt < NKT; ++kt) {
      int klo = kt * 32 + khalf;
      f[kt].q[0] = *(const uint4*)(xbase + klo * 2);
      f[kt].q[1] = *(const uint4*)(xbase + (klo + 16) * 2);
    }
  };

  // One LSTM time step: consume axu (x_t fragments), preload axp with x_{t+1}.
  auto step = [&](int t, Frag32 (&axu)[NKT], Frag32 (&axp)[NKT]) {
    const int wb = t & 1, rb = wb ^ 1;

    // ---- 1. issue x_{t+1} fragment loads (consumed next step, after barrier) ----
    load_x(axp, t + 1);   // one-step overrun at t==T-1 stays inside workspace

    // ---- 2. recurrent: acc = h_{t-1} @ W_hh^T + bias  (W_hh in VGPRs) ----
    v8f acc[4];
    const char* hbase = s_h + rb * 16 * H_STRIDE + ln * H_STRIDE;
#pragma unroll
    for (int kt = 0; kt < 4; ++kt) {
      Frag32 ah;
      int klo = kt * 32 + khalf;
      ah.q[0] = *(const uint4*)(hbase + klo * 2);
      ah.q[1] = *(const uint4*)(hbase + (klo + 16) * 2);
#pragma unroll
      for (int j = 0; j < 4; ++j) {
        // first WMMA reads loop-invariant bias vector as C (no acc-init movs)
        acc[j] = __builtin_amdgcn_wmma_f32_16x16x32_bf16(false, ah.v, false, bhh[j][kt].v,
                                                         (short)0, kt == 0 ? bias_v[j] : acc[j],
                                                         false, false);
      }
    }

    // ---- 3. projection: acc += x_t @ W_ih^T  (W_ih fragments from LDS).
    //      All 4 B-fragments of a k-tile are loaded before the first WMMA so
    //      the 8 ds_loads are in flight together (deep partial waits). ----
#pragma unroll
    for (int kt = 0; kt < NKT; ++kt) {
      int klo = kt * 32 + khalf;
      Frag32 bx[4];
#pragma unroll
      for (int j = 0; j < 4; ++j) {
        const char* wrow = s_wih + ((wv + 8 * j) * 16 + ln) * WIH_STRIDE;
        bx[j].q[0] = *(const uint4*)(wrow + klo * 2);
        bx[j].q[1] = *(const uint4*)(wrow + (klo + 16) * 2);
      }
#pragma unroll
      for (int j = 0; j < 4; ++j)
        acc[j] = __builtin_amdgcn_wmma_f32_16x16x32_bf16(false, axu[kt].v, false, bx[j].v,
                                                         (short)0, acc[j], false, false);
    }

    // ---- 4. LSTM pointwise, entirely in accumulator layout ----
    char* hw = s_h + wb * 16 * H_STRIDE;
    const int ncol = wv * 16 + ln;
#pragma unroll
    for (int k = 0; k < 8; ++k) {
      float ig = fast_sigmoid(acc[0][k]);
      float fg = fast_sigmoid(acc[1][k]);
      float gg = fast_tanh  (acc[2][k]);
      float og = fast_sigmoid(acc[3][k]);
      float cn = fg * c[k] + ig * gg;
      c[k] = cn;
      float hv = og * fast_tanh(cn);
      int m = k + khalf;                       // C/D layout: M = k + 8*(lane>=16)
      *(__bf16*)(hw + m * H_STRIDE + ncol * 2) = (__bf16)hv;
    }

    __syncthreads();   // publish h_t; previous-buffer reads all retired

    // ---- 5. coalesced h tile -> global (one b128 per thread) ----
    {
      int row = tid >> 4, ch = tid & 15;
      uint4 d = *(const uint4*)(hw + row * H_STRIDE + ch * 16);
      *(uint4*)((char*)(hout + (((size_t)(b0 + row)) * T + t) * H) + ch * 16) = d;
    }
  };

  __syncthreads();

  Frag32 axA[NKT], axB[NKT];
  load_x(axA, 0);
  for (int t = 0; t < T; t += 2) {
    step(t,     axA, axB);   // uses x_t,    preloads x_{t+1}
    step(t + 1, axB, axA);   // uses x_{t+1}, preloads x_{t+2}
  }
}

// ---------------------------------------------------------------------------
// Head: out[b] = sigmoid( gelu(last @ w1^T + bb1) @ w2^T + bb2 )
// ---------------------------------------------------------------------------
__global__ void head_kernel(const __bf16* __restrict__ h2, const float* __restrict__ w1,
                            const float* __restrict__ bb1, const float* __restrict__ w2,
                            const float* __restrict__ bb2, float* __restrict__ out, int T) {
  __shared__ float z[64];
  int b = blockIdx.x, j = threadIdx.x;
  const __bf16* last = h2 + ((size_t)b * T + (T - 1)) * 128;
  float a = bb1[j];
  for (int k = 0; k < 128; ++k) a += (float)last[k] * w1[j * 128 + k];
  float g = 0.5f * a * (1.0f + erff(a * 0.70710678f));
  z[j] = g * w2[j];
  __syncthreads();
  if (j == 0) {
    float s = bb2[0];
    for (int k = 0; k < 64; ++k) s += z[k];
    out[b] = 1.0f / (1.0f + __expf(-s));
  }
}

// ---------------------------------------------------------------------------
extern "C" void kernel_launch(void* const* d_in, const int* in_sizes, int n_in,
                              void* d_out, int out_size, void* d_ws, size_t ws_size,
                              hipStream_t stream) {
  (void)in_sizes; (void)n_in; (void)out_size; (void)ws_size;
  const int B = 256, T = 1024, H = 128, NG = 512;

  const float* x = (const float*)d_in[0];
  const float* w_ih[3] = {(const float*)d_in[1], (const float*)d_in[5], (const float*)d_in[9]};
  const float* w_hh[3] = {(const float*)d_in[2], (const float*)d_in[6], (const float*)d_in[10]};
  const float* b_ih[3] = {(const float*)d_in[3], (const float*)d_in[7], (const float*)d_in[11]};
  const float* b_hh[3] = {(const float*)d_in[4], (const float*)d_in[8], (const float*)d_in[12]};
  const float* w1  = (const float*)d_in[13];
  const float* bb1 = (const float*)d_in[14];
  const float* w2  = (const float*)d_in[15];
  const float* bb2 = (const float*)d_in[16];
  float* out = (float*)d_out;

  char* p = (char*)d_ws;
  auto carve = [&](size_t bytes) { char* r = p; p += (bytes + 255) & ~(size_t)255; return r; };
  __bf16* x_bf = (__bf16*)carve((size_t)B * T * 32 * 2);
  __bf16* hb0  = (__bf16*)carve((size_t)B * T * H * 2);
  __bf16* hb1  = (__bf16*)carve((size_t)B * T * H * 2);
  __bf16* wihb[3]; __bf16* whhb[3]; float* bsum[3];
  const int kin[3] = {32, H, H};
  for (int l = 0; l < 3; ++l) {
    wihb[l] = (__bf16*)carve((size_t)NG * kin[l] * 2);
    whhb[l] = (__bf16*)carve((size_t)NG * H * 2);
    bsum[l] = (float*)carve((size_t)NG * 4);
  }

  cvt_f32_bf16_kernel<<<1024, 256, 0, stream>>>(x, x_bf, B * T * 32);
  for (int l = 0; l < 3; ++l) {
    cvt_f32_bf16_kernel<<<64, 256, 0, stream>>>(w_ih[l], wihb[l], NG * kin[l]);
    cvt_f32_bf16_kernel<<<64, 256, 0, stream>>>(w_hh[l], whhb[l], NG * H);
    vec_add_kernel<<<2, 256, 0, stream>>>(b_ih[l], b_hh[l], bsum[l], NG);
  }

  const size_t lds_h  = 2 * 16 * (H * 2 + 16);
  const size_t lds32  = (size_t)NG * (32 * 2 + 16) + lds_h;    // 49,664 B
  const size_t lds128 = (size_t)NG * (128 * 2 + 16) + lds_h;   // 147,968 B

  lstm_layer_kernel<32> <<<B / 16, 256, lds32,  stream>>>(x_bf, wihb[0], whhb[0], bsum[0], hb0, T);
  lstm_layer_kernel<128><<<B / 16, 256, lds128, stream>>>(hb0,  wihb[1], whhb[1], bsum[1], hb1, T);
  lstm_layer_kernel<128><<<B / 16, 256, lds128, stream>>>(hb1,  wihb[2], whhb[2], bsum[2], hb0, T);

  head_kernel<<<B, 64, 0, stream>>>(hb0, w1, bb1, w2, bb2, out, T);
}
